// BoundaryLoss_72112500900190
// MI455X (gfx1250) — compile-verified, matrix operand order checked by reading, not run on hardware
//
#include <hip/hip_runtime.h>
#include <hip/hip_bf16.h>

// Problem dims (fixed by the reference): B=2, D=64, H=96, W=96
#define BB 2
#define DD 64
#define HH 96
#define WW 96
#define NVOX (BB * DD * HH * WW)        // 1,179,648
#define BIGF 1.0e10f
#define NPART (3 * HH * BB)             // blocks in the D-pass kernel: 3*96*2 = 576

typedef float    v2f  __attribute__((ext_vector_type(2)));
typedef float    v8f  __attribute__((ext_vector_type(8)));
typedef _Float16 v16h __attribute__((ext_vector_type(16)));
typedef int      v4i  __attribute__((ext_vector_type(4)));
typedef __attribute__((address_space(1))) v4i as1_v4i;   // global int4
typedef __attribute__((address_space(3))) v4i as3_v4i;   // LDS int4

// ---------------------------------------------------------------------------
// Wave-wide sum using WMMA as a ones-matmul row-sum (CDNA5 matrix pipe).
// A (16x4 f32): VGPR0 = per-lane partial -> A[M=lane%16][K = lane<16 ? 0 : 2],
// VGPR1 = 0. B = all-ones => D[M][N] = p_M + p_{M+16} for every N.
// Each lane sums its 8 D components: lanes 0..15 hold sum_{M=0..7},
// lanes 16..31 hold sum_{M=8..15}; one xor-16 shuffle completes the total.
// ---------------------------------------------------------------------------
__device__ __forceinline__ float wave_reduce_sum(float p) {
#if __has_builtin(__builtin_amdgcn_wmma_f32_16x16x4_f32)
    v2f a;  a[0] = p;    a[1] = 0.0f;
    v2f bb; bb[0] = 1.0f; bb[1] = 1.0f;
    v8f c = {0.f, 0.f, 0.f, 0.f, 0.f, 0.f, 0.f, 0.f};
    c = __builtin_amdgcn_wmma_f32_16x16x4_f32(false, a, false, bb,
                                              (short)0, c, false, false);
    float s = c[0] + c[1] + c[2] + c[3] + c[4] + c[5] + c[6] + c[7];
    s += __shfl_xor(s, 16, 32);
    return s;
#elif __has_builtin(__builtin_amdgcn_wmma_f32_16x16x32_f16)
    // Fallback: same trick with the probe-confirmed f16 WMMA (A[0]=p, B=ones).
    v16h a;  for (int k = 0; k < 16; ++k) a[k]  = (_Float16)0.0f;
    v16h bb; for (int k = 0; k < 16; ++k) bb[k] = (_Float16)1.0f;
    a[0] = (_Float16)p;
    v8f c = {0.f, 0.f, 0.f, 0.f, 0.f, 0.f, 0.f, 0.f};
    c = __builtin_amdgcn_wmma_f32_16x16x32_f16(false, a, false, bb,
                                               (short)0, c, false, false);
    float s = c[0] + c[1] + c[2] + c[3] + c[4] + c[5] + c[6] + c[7];
    s += __shfl_xor(s, 16, 32);
    return s;
#else
    for (int off = 16; off > 0; off >>= 1) p += __shfl_xor(p, off, 32);
    return p;
#endif
}

// ---------------------------------------------------------------------------
// Kernel 1: fused init + W-axis pass.
// One block (96 threads = 3 waves) per contiguous line of 96 floats.
// Wave 0 stages the line into LDS with async global->LDS B128 transfers.
// ---------------------------------------------------------------------------
__global__ void edt_w_kernel(const float* __restrict__ target,
                             float* __restrict__ fp, float* __restrict__ fn) {
    __shared__ __align__(16) float t[WW];
    const int    i    = threadIdx.x;                 // 0..95 (position in line)
    const size_t base = (size_t)blockIdx.x * WW;     // line start

#if __has_builtin(__builtin_amdgcn_global_load_async_to_lds_b128)
    if (i < WW / 4) {   // lanes 0..23 of wave 0 each move 16 bytes
        as1_v4i* g = (as1_v4i*)(target + base + i * 4);
        as3_v4i* l = (as3_v4i*)&t[i * 4];
        __builtin_amdgcn_global_load_async_to_lds_b128(g, l, 0, 0);
    }
  #if __has_builtin(__builtin_amdgcn_s_wait_asynccnt)
    __builtin_amdgcn_s_wait_asynccnt(0);
  #else
    asm volatile("s_wait_asynccnt 0" ::: "memory");
  #endif
#else
    t[i] = target[base + i];
#endif
    __syncthreads();

    float mp = 3.0e38f, mn = 3.0e38f;
    #pragma unroll 4
    for (int j = 0; j < WW; ++j) {
        const float d2 = (float)((i - j) * (i - j));   // exact small integer
        const float tv = t[j];
        const float fpv = (tv > 0.5f) ? BIGF : 0.0f;
        mp = fminf(mp, d2 + fpv);
        mn = fminf(mn, d2 + (BIGF - fpv));             // inverse mask
    }
    fp[base + i] = mp;
    fn[base + i] = mn;
}

// ---------------------------------------------------------------------------
// Kernel 2: H-axis pass, in-place on fp/fn.
// Block (32,8): 32 consecutive w for coalescing x full H line in LDS.
// LDS layout s[h][tx] -> lane-consecutive banks, conflict-free.
// ---------------------------------------------------------------------------
__global__ void edt_h_kernel(float* __restrict__ fp, float* __restrict__ fn) {
    __shared__ float sp[HH][32];
    __shared__ float sn[HH][32];
    const int tx = threadIdx.x, ty = threadIdx.y;
    const int w0 = blockIdx.x * 32;
    const int d  = blockIdx.y, b = blockIdx.z;
    const size_t base = ((size_t)(b * DD + d) * HH) * WW + w0; // (b,d,h=0,w0)

    for (int h = ty; h < HH; h += 8) {
        sp[h][tx] = fp[base + (size_t)h * WW + tx];
        sn[h][tx] = fn[base + (size_t)h * WW + tx];
    }
    __syncthreads();

    for (int h = ty; h < HH; h += 8) {
        float mp = 3.0e38f, mn = 3.0e38f;
        #pragma unroll 4
        for (int j = 0; j < HH; ++j) {
            const float d2 = (float)((h - j) * (h - j));
            mp = fminf(mp, d2 + sp[j][tx]);
            mn = fminf(mn, d2 + sn[j][tx]);
        }
        fp[base + (size_t)h * WW + tx] = mp;
        fn[base + (size_t)h * WW + tx] = mn;
    }
}

// ---------------------------------------------------------------------------
// Kernel 3: D-axis pass fused with weight + BCE + block reduction.
// Block (32,8) covers 32 w x 64 d at fixed (b,h). WMMA wave reduction,
// deterministic per-block partials (no atomics).
// ---------------------------------------------------------------------------
__global__ void edt_d_loss_kernel(const float* __restrict__ pred,
                                  const float* __restrict__ target,
                                  const float* __restrict__ fp,
                                  const float* __restrict__ fn,
                                  float* __restrict__ partials) {
    __shared__ float sp[DD][32];
    __shared__ float sn[DD][32];
    __shared__ float wsum[8][2];
    const int tx = threadIdx.x, ty = threadIdx.y;
    const int w0 = blockIdx.x * 32;
    const int h  = blockIdx.y, b = blockIdx.z;
    const size_t base = (size_t)b * (DD * HH * WW) + (size_t)h * WW + w0;

    for (int d = ty; d < DD; d += 8) {
        sp[d][tx] = fp[base + (size_t)d * (HH * WW) + tx];
        sn[d][tx] = fn[base + (size_t)d * (HH * WW) + tx];
    }
    __syncthreads();

    float np = 0.0f, dp = 0.0f;
    for (int d = ty; d < DD; d += 8) {
        float mp = 3.0e38f, mn = 3.0e38f;
        #pragma unroll 4
        for (int j = 0; j < DD; ++j) {
            const float d2 = (float)((d - j) * (d - j));
            mp = fminf(mp, d2 + sp[j][tx]);
            mn = fminf(mn, d2 + sn[j][tx]);
        }
        const float a = fabsf(sqrtf(mn) - sqrtf(mp));
        const float wv = (a <= 3.0f) ? 1.0f
                        : ((a >= 5.0f) ? 0.0f : 1.0f - (a - 3.0f) * 0.5f);
        const size_t idx = base + (size_t)d * (HH * WW) + tx;
        const float p  = pred[idx];
        const float tg = target[idx];
        const float lp   = fmaxf(logf(p),        -100.0f);
        const float l1mp = fmaxf(logf(1.0f - p), -100.0f);
        const float bce  = -(tg * lp + (1.0f - tg) * l1mp);
        np += bce * wv;
        dp += wv;
    }

    // Wave (32-lane) reduction on the matrix pipe; EXEC is all-ones here.
    const float num_w = wave_reduce_sum(np);
    const float den_w = wave_reduce_sum(dp);
    if (tx == 0) { wsum[ty][0] = num_w; wsum[ty][1] = den_w; }
    __syncthreads();

    if (tx == 0 && ty == 0) {
        float nsum = 0.0f, dsum = 0.0f;
        #pragma unroll
        for (int k = 0; k < 8; ++k) { nsum += wsum[k][0]; dsum += wsum[k][1]; }
        const int blk = (blockIdx.z * HH + blockIdx.y) * 3 + blockIdx.x; // b-major
        partials[2 * blk + 0] = nsum;
        partials[2 * blk + 1] = dsum;
    }
}

// ---------------------------------------------------------------------------
// Kernel 4: deterministic fixed-order final combine (576 partial pairs).
// ---------------------------------------------------------------------------
__global__ void finalize_kernel(const float* __restrict__ partials,
                                float* __restrict__ out) {
    if (threadIdx.x == 0 && blockIdx.x == 0) {
        const int per_b = NPART / BB;  // 288
        float acc = 0.0f;
        for (int b = 0; b < BB; ++b) {
            float nsum = 0.0f, dsum = 0.0f;
            for (int k = 0; k < per_b; ++k) {
                nsum += partials[2 * (b * per_b + k) + 0];
                dsum += partials[2 * (b * per_b + k) + 1];
            }
            acc += nsum / (dsum + 1e-5f);
        }
        out[0] = acc / (float)BB;
    }
}

// ---------------------------------------------------------------------------
extern "C" void kernel_launch(void* const* d_in, const int* in_sizes, int n_in,
                              void* d_out, int out_size, void* d_ws, size_t ws_size,
                              hipStream_t stream) {
    const float* pred   = (const float*)d_in[0];
    const float* target = (const float*)d_in[1];
    float* out = (float*)d_out;

    float* fp       = (float*)d_ws;
    float* fn       = fp + NVOX;
    float* partials = fn + NVOX;
    if (ws_size < (size_t)(2 * NVOX + 2 * NPART) * sizeof(float)) return;

    // Pass 1: init from target + W-axis transform (contiguous lines, async LDS).
    edt_w_kernel<<<dim3(BB * DD * HH), dim3(WW), 0, stream>>>(target, fp, fn);
    // Pass 2: H-axis transform (in-place).
    edt_h_kernel<<<dim3(WW / 32, DD, BB), dim3(32, 8), 0, stream>>>(fp, fn);
    // Pass 3: D-axis transform fused with weight/BCE + WMMA reduction.
    edt_d_loss_kernel<<<dim3(WW / 32, HH, BB), dim3(32, 8), 0, stream>>>(
        pred, target, fp, fn, partials);
    // Pass 4: deterministic scalar combine.
    finalize_kernel<<<dim3(1), dim3(32), 0, stream>>>(partials, out);
}
